// ScaledEuclidDistanceAttention_18305150616106
// MI455X (gfx1250) — matrix operand order repaired and checked by reading, not compiled, not run on hardware
//
#include <hip/hip_runtime.h>

// ---------------------------------------------------------------------------
// CDNA5 (gfx1250) wave32 WMMA implementation of scaled-Euclidean-distance
// multi-head attention.  Matmuls: v_wmma_f32_16x16x32_bf16.
// K/V tiles: double-buffered global_load_async_to_lds_b128 (ASYNCcnt).
// ---------------------------------------------------------------------------

typedef __attribute__((ext_vector_type(16))) __bf16 bf16x16;
typedef __attribute__((ext_vector_type(8)))  __bf16 bf16x8;
typedef __attribute__((ext_vector_type(8)))  float  f32x8;

static constexpr int BB = 2;
static constexpr int SS = 2048;
static constexpr int DD = 1024;
static constexpr int HH = 16;
static constexpr int DH = 64;   // head dim

__device__ __forceinline__ bf16x8 load8_cvt(const float* p) {
    float4 a = *reinterpret_cast<const float4*>(p);
    float4 b = *reinterpret_cast<const float4*>(p + 4);
    bf16x8 r;
    r[0] = (__bf16)a.x; r[1] = (__bf16)a.y; r[2] = (__bf16)a.z; r[3] = (__bf16)a.w;
    r[4] = (__bf16)b.x; r[5] = (__bf16)b.y; r[6] = (__bf16)b.z; r[7] = (__bf16)b.w;
    return r;
}

__device__ __forceinline__ bf16x16 combine(bf16x8 lo, bf16x8 hi) {
    union { bf16x16 v; bf16x8 h[2]; } u;
    u.h[0] = lo; u.h[1] = hi;
    return u.v;
}

__device__ __forceinline__ f32x8 wmma_bf16(bf16x16 a, bf16x16 b, f32x8 c) {
    return __builtin_amdgcn_wmma_f32_16x16x32_bf16(false, a, false, b,
                                                   (short)0, c, false, false);
}

// A fragment, 16x32 bf16 tile (row-major source).  lane row = lane&15,
// K packing: half 0 -> {0..15}, half 1 -> {16..31} (contig).
__device__ __forceinline__ bf16x16 a_frag_bf16(const __bf16* base32, int half) {
    const __bf16* p = base32 + half * 16;
    return combine(*reinterpret_cast<const bf16x8*>(p),
                   *reinterpret_cast<const bf16x8*>(p + 8));
}

// B fragment, 32x16 bf16 tile: lane col = lane&15, K = 16*half + j (contig).
__device__ __forceinline__ bf16x16 b_frag_bf16(const __bf16* base32, int half) {
    const __bf16* p = base32 + half * 16;
    return combine(*reinterpret_cast<const bf16x8*>(p),
                   *reinterpret_cast<const bf16x8*>(p + 8));
}

// Async global -> LDS copy, 16B per lane, tracked by ASYNCcnt.
__device__ __forceinline__ void async_load_b128(unsigned lds_off, const void* gptr) {
    asm volatile("global_load_async_to_lds_b128 %0, %1, off"
                 :: "v"(lds_off), "v"(gptr) : "memory");
}

// ---------------------------------------------------------------------------
// Elementwise f32 -> bf16 conversion (8 elements / thread, 16B stores)
// ---------------------------------------------------------------------------
__global__ __launch_bounds__(256)
void cvt_kernel(const float* __restrict__ src, __bf16* __restrict__ dst, int n8) {
    const int i = blockIdx.x * blockDim.x + threadIdx.x;
    if (i < n8) {
        *reinterpret_cast<bf16x8*>(dst + (size_t)i * 8) =
            load8_cvt(src + (size_t)i * 8);
    }
}

// ---------------------------------------------------------------------------
// V transpose: Vp[b*S+s][D] -> Vt[((b*H+h)*DH+d)*S + s]   (bf16)
// grid = (S/64, D/64, B), block = 256.  64x64 tiles staged in LDS.
// ---------------------------------------------------------------------------
__global__ __launch_bounds__(256)
void transpose_v(const __bf16* __restrict__ Vp, __bf16* __restrict__ Vt) {
    constexpr int ST = 72;  // padded LDS row stride (16B aligned)
    __shared__ __bf16 tile[64 * ST];
    const int tid = threadIdx.x;
    const int s0  = blockIdx.x * 64;
    const size_t bbase = (size_t)blockIdx.z * SS;
#pragma unroll
    for (int i = 0; i < 2; ++i) {
        const int cid = tid * 2 + i;      // 0..511
        const int row = cid >> 3;         // s within tile
        const int c8  = (cid & 7) * 8;    // dim offset
        *reinterpret_cast<bf16x8*>(&tile[row * ST + c8]) =
            *reinterpret_cast<const bf16x8*>(
                Vp + (bbase + s0 + row) * DD + blockIdx.y * 64 + c8);
    }
    __syncthreads();
#pragma unroll
    for (int i = 0; i < 2; ++i) {
        const int cid = tid * 2 + i;
        const int d   = cid >> 3;         // dim within tile (= dh, head tile)
        const int s8  = (cid & 7) * 8;    // s offset
        bf16x8 vv;
#pragma unroll
        for (int e = 0; e < 8; ++e) vv[e] = tile[(s8 + e) * ST + d];
        const size_t dst =
            ((size_t)(blockIdx.z * HH + blockIdx.y) * DH + d) * SS + s0 + s8;
        *reinterpret_cast<bf16x8*>(Vt + dst) = vv;
    }
}

// ---------------------------------------------------------------------------
// GEMM:  Y[M,N] = A[M,K] @ W[N,K]^T   (bf16 operands)
//   FINAL=false: bf16 out, optional per-(row,head) sum-of-squares -> norms
//   FINAL=true : f32 out + bias
// block = 128 threads (4 waves).  Block tile 128x64, wave tile 32x64.
// ---------------------------------------------------------------------------
template <bool FINAL>
__global__ __launch_bounds__(128)
void gemm_bf16(const __bf16* __restrict__ A,
               const __bf16* __restrict__ W,
               const float* __restrict__ bias,
               void* __restrict__ Yopaque,
               float* __restrict__ norms,
               int M, int N, int K) {
    const int lane  = threadIdx.x & 31;
    const int w     = threadIdx.x >> 5;
    const int half  = lane >> 4;
    const int l16   = lane & 15;
    const int rbase = blockIdx.y * 128 + w * 32;
    const int cbase = blockIdx.x * 64;

    f32x8 acc[2][4] = {};

    for (int kc = 0; kc < K; kc += 32) {
        bf16x16 a[2];
#pragma unroll
        for (int g = 0; g < 2; ++g)
            a[g] = a_frag_bf16(A + (size_t)(rbase + g * 16 + l16) * K + kc, half);
#pragma unroll
        for (int u = 0; u < 4; ++u) {
            bf16x16 b = b_frag_bf16(W + (size_t)(cbase + u * 16 + l16) * K + kc, half);
            acc[0][u] = wmma_bf16(a[0], b, acc[0][u]);
            acc[1][u] = wmma_bf16(a[1], b, acc[1][u]);
        }
    }

#pragma unroll
    for (int g = 0; g < 2; ++g)
#pragma unroll
        for (int u = 0; u < 4; ++u)
#pragma unroll
            for (int r = 0; r < 8; ++r) {
                const int row = rbase + g * 16 + r + 8 * half;
                const int col = cbase + u * 16 + l16;
                if constexpr (FINAL) {
                    ((float*)Yopaque)[(size_t)row * N + col] = acc[g][u][r] + bias[col];
                } else {
                    ((__bf16*)Yopaque)[(size_t)row * N + col] = (__bf16)acc[g][u][r];
                }
            }

    if constexpr (!FINAL) {
        if (norms != nullptr) {
            const int h = blockIdx.x;  // one head per 64-wide col tile
#pragma unroll
            for (int g = 0; g < 2; ++g)
#pragma unroll
                for (int r = 0; r < 8; ++r) {
                    float s = 0.f;
#pragma unroll
                    for (int u = 0; u < 4; ++u) {
                        const float vb = (float)(__bf16)acc[g][u][r];
                        s += vb * vb;
                    }
                    s += __shfl_xor(s, 1);
                    s += __shfl_xor(s, 2);
                    s += __shfl_xor(s, 4);
                    s += __shfl_xor(s, 8);
                    if (l16 == 0) {
                        const int row = rbase + g * 16 + r + 8 * half;
                        norms[(size_t)row * HH + h] = s;
                    }
                }
        }
    }
}

// ---------------------------------------------------------------------------
// Fused flash-style Euclidean-distance attention.
// grid = (S/128, H, B), block = 256 threads (8 waves, 16 q-rows each).
// 64 keys / iteration; K and V(transposed) tiles double-buffered via
// global_load_async_to_lds_b128.
// ---------------------------------------------------------------------------
__device__ __forceinline__ void issue_tile_loads(const __bf16* Kbase,
                                                 const __bf16* Vbase,
                                                 __bf16* ksbuf, __bf16* vbuf,
                                                 int tid) {
#pragma unroll
    for (int i = 0; i < 2; ++i) {
        const int cid = tid * 2 + i;      // 0..511
        const int row = cid >> 3;         // key (K tile) / dim (V tile)
        const int c8  = (cid & 7) * 8;
        async_load_b128((unsigned)(size_t)&ksbuf[row * 64 + c8],
                        Kbase + (size_t)row * DD + c8);
        async_load_b128((unsigned)(size_t)&vbuf[row * 64 + c8],
                        Vbase + (size_t)row * SS + c8);
    }
}

__global__ __launch_bounds__(256)
void attn_kernel(const __bf16* __restrict__ Qp,
                 const __bf16* __restrict__ Kp,
                 const __bf16* __restrict__ Vt,
                 const float* __restrict__ normsQ,
                 const float* __restrict__ normsK,
                 __bf16* __restrict__ Oh) {
    constexpr float scale = 0.125f;  // dh^-0.5 = 1/8

    __shared__ __bf16 ks2[2][64 * 64];   // K tiles  [key][dim]   2 x 8 KB
    __shared__ __bf16 vs2[2][64 * 64];   // V tiles  [dim][key]   2 x 8 KB
    __shared__ __bf16 ps[8][16 * 64];    // per-wave P staging    16 KB

    const int tid  = threadIdx.x;
    const int lane = tid & 31;
    const int w    = tid >> 5;
    const int half = lane >> 4;
    const int l16  = lane & 15;

    const int    h     = blockIdx.y;
    const size_t brow  = (size_t)blockIdx.z * SS;
    const int    qbase = blockIdx.x * 128 + w * 16;
    const int    hcol  = h * DH;

    const __bf16* Kblk = Kp + brow * DD + hcol;
    const __bf16* Vblk = Vt + (size_t)(blockIdx.z * HH + h) * DH * SS;

    // --- Q fragments (resident for the whole key loop) ---
    bf16x16 aq[2];
    {
        const __bf16* qrow = Qp + (brow + qbase + l16) * DD + hcol;
#pragma unroll
        for (int c = 0; c < 2; ++c) aq[c] = a_frag_bf16(qrow + c * 32, half);
    }
    float q2v[8];
#pragma unroll
    for (int r = 0; r < 8; ++r)
        q2v[r] = normsQ[(brow + qbase + r + 8 * half) * HH + h];

    f32x8 o[4] = {};
    float lsum[8], mrow[8];
#pragma unroll
    for (int r = 0; r < 8; ++r) { lsum[r] = 0.f; mrow[r] = -1e30f; }

    int buf = 0;
    issue_tile_loads(Kblk, Vblk, ks2[0], vs2[0], tid);  // prime tile 0

    for (int kb = 0; kb < SS; kb += 64) {
        const bool more = (kb + 64) < SS;
        if (more)
            issue_tile_loads(Kblk + (size_t)(kb + 64) * DD, Vblk + (kb + 64),
                             ks2[buf ^ 1], vs2[buf ^ 1], tid);
        // ||k||^2 for this lane's key column in each 16-key subtile
        float k2[4];
#pragma unroll
        for (int t = 0; t < 4; ++t)
            k2[t] = normsK[(brow + kb + t * 16 + l16) * HH + h];

        if (more) asm volatile("s_wait_asynccnt 4" ::: "memory");
        else      asm volatile("s_wait_asynccnt 0" ::: "memory");
        __syncthreads();

        const __bf16* ksb = ks2[buf];
        const __bf16* vsb = vs2[buf];

        // scores: S = Q @ K^T  (four 16-key subtiles, K-dim 64 = 2 chunks)
        f32x8 sc[4];
#pragma unroll
        for (int t = 0; t < 4; ++t) {
            bf16x16 b0 = b_frag_bf16(&ksb[(t * 16 + l16) * 64], half);
            bf16x16 b1 = b_frag_bf16(&ksb[(t * 16 + l16) * 64 + 32], half);
            f32x8 s = {};
            s = wmma_bf16(aq[0], b0, s);
            s = wmma_bf16(aq[1], b1, s);
            sc[t] = s;
        }

        float alpha[8];
#pragma unroll
        for (int r = 0; r < 8; ++r) {
            float x[4];
#pragma unroll
            for (int t = 0; t < 4; ++t) {
                const float d2 = fmaxf(q2v[r] + k2[t] - 2.f * sc[t][r], 0.f);
                x[t] = __builtin_amdgcn_rcpf(
                    __builtin_amdgcn_sqrtf(d2) * scale + 1e-9f);
            }
            float loc = fmaxf(fmaxf(x[0], x[1]), fmaxf(x[2], x[3]));
            loc = fmaxf(loc, __shfl_xor(loc, 1));
            loc = fmaxf(loc, __shfl_xor(loc, 2));
            loc = fmaxf(loc, __shfl_xor(loc, 4));
            loc = fmaxf(loc, __shfl_xor(loc, 8));
            const float mnew = fmaxf(mrow[r], loc);
            alpha[r] = __expf(mrow[r] - mnew);
            mrow[r]  = mnew;
            float p[4], rs = 0.f;
#pragma unroll
            for (int t = 0; t < 4; ++t) { p[t] = __expf(x[t] - mnew); rs += p[t]; }
            rs += __shfl_xor(rs, 1);
            rs += __shfl_xor(rs, 2);
            rs += __shfl_xor(rs, 4);
            rs += __shfl_xor(rs, 8);
            lsum[r] = lsum[r] * alpha[r] + rs;
            const int rl = r + 8 * half;
#pragma unroll
            for (int t = 0; t < 4; ++t)
                ps[w][rl * 64 + t * 16 + l16] = (__bf16)p[t];
        }
#pragma unroll
        for (int u = 0; u < 4; ++u)
#pragma unroll
            for (int r = 0; r < 8; ++r) o[u][r] *= alpha[r];

        // ps is per-wave private; DS ops are in-order within a wave, so the
        // A-layout re-read below needs no block barrier.
        bf16x16 ap0 = a_frag_bf16(&ps[w][l16 * 64], half);
        bf16x16 ap1 = a_frag_bf16(&ps[w][l16 * 64 + 32], half);
#pragma unroll
        for (int u = 0; u < 4; ++u) {
            bf16x16 bv0 = b_frag_bf16(&vsb[(u * 16 + l16) * 64], half);
            bf16x16 bv1 = b_frag_bf16(&vsb[(u * 16 + l16) * 64 + 32], half);
            o[u] = wmma_bf16(ap0, bv0, o[u]);
            o[u] = wmma_bf16(ap1, bv1, o[u]);
        }

        __syncthreads();   // all reads of buf done before it is refilled
        buf ^= 1;
    }

    // epilogue: normalize and store per-head output (bf16)
#pragma unroll
    for (int r = 0; r < 8; ++r) lsum[r] = __builtin_amdgcn_rcpf(lsum[r]);
#pragma unroll
    for (int u = 0; u < 4; ++u)
#pragma unroll
        for (int r = 0; r < 8; ++r) {
            const size_t row = brow + qbase + r + 8 * half;
            const int    col = hcol + u * 16 + l16;
            Oh[row * DD + col] = (__bf16)(o[u][r] * lsum[r]);
        }
}

// ---------------------------------------------------------------------------
extern "C" void kernel_launch(void* const* d_in, const int* in_sizes, int n_in,
                              void* d_out, int out_size, void* d_ws, size_t ws_size,
                              hipStream_t stream) {
    const float* q     = (const float*)d_in[0];
    const float* k     = (const float*)d_in[1];
    const float* v     = (const float*)d_in[2];
    const float* Wq    = (const float*)d_in[3];
    const float* Wk    = (const float*)d_in[4];
    const float* Wv    = (const float*)d_in[5];
    const float* Wout  = (const float*)d_in[6];
    const float* b_out = (const float*)d_in[7];

    const int    M  = BB * SS;          // 4096 token rows
    const size_t MD = (size_t)M * DD;   // elements per activation buffer
    const size_t WW = (size_t)DD * DD;  // elements per weight matrix

    __bf16* xb  = (__bf16*)d_ws;        // staging bf16 q/k/v; later reused as Vt
    __bf16* Wqb = xb + MD;
    __bf16* Wkb = Wqb + WW;
    __bf16* Wvb = Wkb + WW;
    __bf16* Wob = Wvb + WW;
    __bf16* Qp  = Wob + WW;             // [M, D] bf16 projections
    __bf16* Kp  = Qp + MD;
    __bf16* Vp  = Kp + MD;
    __bf16* Oh  = Vp + MD;              // per-head attention output
    float*  nQ  = (float*)(Oh + MD);    // [M, H] ||q_h||^2
    float*  nK  = nQ + (size_t)M * HH;  // [M, H] ||k_h||^2
    __bf16* Vt  = xb;                   // transposed V, aliases dead staging

    const dim3 blk(128);
    const dim3 gg(DD / 64, M / 128);
    const int  nx8 = (int)(MD / 8);
    const int  nw8 = (int)(WW / 8);
    const int  cvtx = (nx8 + 255) / 256;
    const int  cvtw = (nw8 + 255) / 256;

    // weights -> bf16 (once each)
    cvt_kernel<<<cvtw, 256, 0, stream>>>(Wq,   Wqb, nw8);
    cvt_kernel<<<cvtw, 256, 0, stream>>>(Wk,   Wkb, nw8);
    cvt_kernel<<<cvtw, 256, 0, stream>>>(Wv,   Wvb, nw8);
    cvt_kernel<<<cvtw, 256, 0, stream>>>(Wout, Wob, nw8);

    // projections (activations staged through one reused bf16 buffer)
    cvt_kernel<<<cvtx, 256, 0, stream>>>(q, xb, nx8);
    gemm_bf16<false><<<gg, blk, 0, stream>>>(xb, Wqb, nullptr, Qp, nQ, M, DD, DD);
    cvt_kernel<<<cvtx, 256, 0, stream>>>(k, xb, nx8);
    gemm_bf16<false><<<gg, blk, 0, stream>>>(xb, Wkb, nullptr, Kp, nK, M, DD, DD);
    cvt_kernel<<<cvtx, 256, 0, stream>>>(v, xb, nx8);
    gemm_bf16<false><<<gg, blk, 0, stream>>>(xb, Wvb, nullptr, Vp, nullptr, M, DD, DD);

    // V -> per-head transposed layout (xb is dead, reuse as Vt)
    transpose_v<<<dim3(SS / 64, DD / 64, BB), 256, 0, stream>>>(Vp, Vt);

    // fused distance attention (async double-buffered K/V tiles)
    attn_kernel<<<dim3(SS / 128, HH, BB), 256, 0, stream>>>(Qp, Kp, Vt, nQ, nK, Oh);

    // output projection + bias
    gemm_bf16<true><<<gg, blk, 0, stream>>>(Oh, Wob, b_out, (float*)d_out, nullptr, M, DD, DD);
}
